// Multi_Head_Attention_3710851744495
// MI455X (gfx1250) — matrix operand order repaired and checked by reading, not compiled, not run on hardware
//
#include <hip/hip_runtime.h>

#define EMBED 256
#define HEADS 4
#define NSEQ  4096
#define BATCH 2
#define DHEAD 64

typedef __attribute__((ext_vector_type(16))) _Float16 v16h;
typedef __attribute__((ext_vector_type(8)))  float    v8f;
typedef __attribute__((ext_vector_type(4)))  int      v4i;

// gfx1250 async global->LDS path (ASYNCcnt tracked).
#if defined(__gfx1250__) &&                                         \
    __has_builtin(__builtin_amdgcn_global_load_async_to_lds_b128) && \
    __has_builtin(__builtin_amdgcn_s_wait_asynccnt)
#define USE_ASYNC_LDS 1
#else
#define USE_ASYNC_LDS 0
#endif

#if USE_ASYNC_LDS
typedef __attribute__((address_space(1))) v4i global_v4i;
typedef __attribute__((address_space(3))) v4i local_v4i;
__device__ __forceinline__ void async16(const _Float16* g, _Float16* l) {
  __builtin_amdgcn_global_load_async_to_lds_b128(
      (global_v4i*)(uintptr_t)g, (local_v4i*)(uintptr_t)l, 0, 0);
}
#endif

// D = A(16x32 f16) * B(32x16 f16) + C(16x16 f32)
__device__ __forceinline__ v8f wmma16(v16h a, v16h b, v8f c) {
  return __builtin_amdgcn_wmma_f32_16x16x32_f16(false, a, false, b, (short)0, c,
                                                false, false);
}

// ---- Fragment loaders -------------------------------------------------------
// A fragment (M=16,K=32): lane holds row = lane&15; half = lane>>4.
// half-slot j -> k = 8*half + j + (j>=8 ? 8 : 0)  => two contiguous runs of 8.
__device__ __forceinline__ v16h a_frag_h16(const _Float16* rowp, int half) {
  union { uint4 u[2]; v16h h; } x;
  const _Float16* p0 = rowp + 8 * half;
  x.u[0] = *(const uint4*)(p0);
  x.u[1] = *(const uint4*)(p0 + 16);
  return x.h;
}
// B fragment (K=32,N=16): lane holds col = lane&15.
// half-slot j -> k = 8*half + j + (j>=8 ? 8 : 0)  => two contiguous runs of 8.
__device__ __forceinline__ v16h b_frag_h16(const _Float16* p, int half) {
  union { uint4 u[2]; v16h h; } x;
  const _Float16* q = p + 8 * half;
  x.u[0] = *(const uint4*)(q);
  x.u[1] = *(const uint4*)(q + 16);
  return x.h;
}

// ---- Stage 0: one-shot fp32 -> f16 conversion (x and the four weights) -----
__global__ __launch_bounds__(256) void cvt_f16_kernel(
    const float* __restrict__ src, _Float16* __restrict__ dst, int n8) {
  const int i = blockIdx.x * blockDim.x + threadIdx.x;
  if (i >= n8) return;
  const float4* s = (const float4*)src + (size_t)i * 2;
  const float4 a = s[0], b = s[1];
  union { uint4 u; _Float16 h[8]; } pk;
  pk.h[0] = (_Float16)a.x; pk.h[1] = (_Float16)a.y;
  pk.h[2] = (_Float16)a.z; pk.h[3] = (_Float16)a.w;
  pk.h[4] = (_Float16)b.x; pk.h[5] = (_Float16)b.y;
  pk.h[6] = (_Float16)b.z; pk.h[7] = (_Float16)b.w;
  ((uint4*)dst)[i] = pk.u;
}

// ---- Stage 1: fused QKV projection -----------------------------------------
// grid = (B*N/32, 3), block = 128 (4 waves). Wave owns 32 rows x 64 cols:
// every B fragment feeds two WMMAs. All operands pre-converted f16.
// mat 0 -> Q[B,H,N,64], mat 1 -> K[B,H,N,64], mat 2 -> Vt[B,H,64,N] (transposed)
__global__ __launch_bounds__(128) void qkv_proj_kernel(
    const _Float16* __restrict__ xh,
    const _Float16* __restrict__ wqh, const float* __restrict__ bq,
    const _Float16* __restrict__ wkh, const float* __restrict__ bk,
    const _Float16* __restrict__ wvh, const float* __restrict__ bv,
    _Float16* __restrict__ Qo, _Float16* __restrict__ Ko,
    _Float16* __restrict__ Vt) {
  const int mat = blockIdx.y;
  const _Float16* w = (mat == 0) ? wqh : (mat == 1) ? wkh : wvh;
  const float* bias = (mat == 0) ? bq : (mat == 1) ? bk : bv;

  const int lane = threadIdx.x & 31;
  const int wid  = threadIdx.x >> 5;
  const int half = lane >> 4;
  const int l16  = lane & 15;

  const int g0 = blockIdx.x * 32;  // 32-row tile over flattened B*N
  const int b  = g0 / NSEQ;
  const int n0 = g0 % NSEQ;

  const _Float16* xrow0 = xh + (size_t)(g0 + l16) * EMBED;
  const _Float16* xrow1 = xrow0 + (size_t)16 * EMBED;
  const int eb = wid * 64;

  v8f acc[2][4] = {};
#pragma unroll
  for (int k0 = 0; k0 < EMBED; k0 += 32) {
    const v16h a0 = a_frag_h16(xrow0 + k0, half);
    const v16h a1 = a_frag_h16(xrow1 + k0, half);
#pragma unroll
    for (int t = 0; t < 4; ++t) {
      // B[k][e] = W[e][k]: lane's column e, k contiguous along W row e.
      const v16h bf = b_frag_h16(w + (size_t)(eb + t * 16 + l16) * EMBED + k0, half);
      acc[0][t] = wmma16(a0, bf, acc[0][t]);
      acc[1][t] = wmma16(a1, bf, acc[1][t]);
    }
  }

#pragma unroll
  for (int mt = 0; mt < 2; ++mt) {
    const int nb8 = n0 + mt * 16 + 8 * half;  // first of this lane's 8 rows
    if (mat < 2) {
      _Float16* out = (mat == 0) ? Qo : Ko;
#pragma unroll
      for (int t = 0; t < 4; ++t) {
        const int e = eb + t * 16 + l16;
        const int h = e >> 6, d = e & 63;
        const float bb = bias[e];
        _Float16* op = out + (((size_t)b * HEADS + h) * NSEQ + nb8) * DHEAD + d;
#pragma unroll
        for (int r = 0; r < 8; ++r)
          op[(size_t)r * DHEAD] = (_Float16)(acc[mt][t][r] + bb);
      }
    } else {
      // V transposed: lane packs its 8 rows (contiguous n) -> one b128 store
#pragma unroll
      for (int t = 0; t < 4; ++t) {
        const int e = eb + t * 16 + l16;
        const int h = e >> 6, d = e & 63;
        const float bb = bias[e];
        union { uint4 u; _Float16 hh[8]; } pk;
#pragma unroll
        for (int r = 0; r < 8; ++r) pk.hh[r] = (_Float16)(acc[mt][t][r] + bb);
        *(uint4*)(Vt + (((size_t)b * HEADS + h) * DHEAD + d) * NSEQ + nb8) = pk.u;
      }
    }
  }
}

// ---- Stage 2: flash attention ----------------------------------------------
// grid = (N/64, B*H), block = 128 (4 waves). Wave owns 16 query rows.
// 64-key blocks double-buffered in LDS via async global->LDS DMA (8 b128 per
// thread in flight); online softmax amortized over 4 S-tiles; 16 WMMAs per
// barrier pair. att = softmax(QK^T)/sqrt(EMBED) -> final scale 1/(l*16).
__global__ __launch_bounds__(128) void flash_attn_kernel(
    const _Float16* __restrict__ Q, const _Float16* __restrict__ K,
    const _Float16* __restrict__ Vt, _Float16* __restrict__ O) {
  __shared__ __attribute__((aligned(16))) _Float16 Ks[2][64 * 64];  // [key][d]
  __shared__ __attribute__((aligned(16))) _Float16 Vs[2][64 * 64];  // [d][key]
  __shared__ __attribute__((aligned(16))) _Float16 Ps[4 * 16 * 64]; // per-wave P

  const int bh   = blockIdx.y;
  const int tid  = threadIdx.x;
  const int lane = tid & 31, wid = tid >> 5;
  const int half = lane >> 4, l16 = lane & 15;
  const int q0 = blockIdx.x * 64 + wid * 16;

  const _Float16* Qrow = Q + ((size_t)bh * NSEQ + q0 + l16) * DHEAD;
  const v16h qf0 = a_frag_h16(Qrow + 0, half);
  const v16h qf1 = a_frag_h16(Qrow + 32, half);

  const _Float16* Kb = K + (size_t)bh * NSEQ * DHEAD;
  const _Float16* Vb = Vt + (size_t)bh * DHEAD * NSEQ;
  _Float16* Pw = Ps + wid * (16 * 64);

  float m[8], l[8];
  v8f o0 = {}, o1 = {}, o2 = {}, o3 = {};
#pragma unroll
  for (int r = 0; r < 8; ++r) { m[r] = -3.0e38f; l[r] = 0.f; }

  // Staging split: 128 threads, 64 rows x 2 half-rows of 32 halves (64B each).
  const int srow = tid >> 1, spart = tid & 1;

  auto stage = [&](int kb, int buf) {
    const _Float16* gk = Kb + ((size_t)(kb + srow)) * DHEAD + spart * 32;
    _Float16* lk = &Ks[buf][srow * 64 + spart * 32];
    const _Float16* gv = Vb + (size_t)srow * NSEQ + kb + spart * 32;
    _Float16* lv = &Vs[buf][srow * 64 + spart * 32];
#if USE_ASYNC_LDS
#pragma unroll
    for (int c = 0; c < 4; ++c) {
      async16(gk + c * 8, lk + c * 8);
      async16(gv + c * 8, lv + c * 8);
    }
#else
#pragma unroll
    for (int c = 0; c < 4; ++c) {
      ((uint4*)lk)[c] = ((const uint4*)gk)[c];
      ((uint4*)lv)[c] = ((const uint4*)gv)[c];
    }
#endif
  };

  constexpr int KB = 64;
  constexpr int NB = NSEQ / KB;
  stage(0, 0);

  for (int i = 0; i < NB; ++i) {
    const int cur = i & 1;
    if (i + 1 < NB) {
      stage((i + 1) * KB, cur ^ 1);  // prefetch next block into other buffer
#if USE_ASYNC_LDS
      __builtin_amdgcn_s_wait_asynccnt(8);  // current block's 8 DMAs done
#endif
    } else {
#if USE_ASYNC_LDS
      __builtin_amdgcn_s_wait_asynccnt(0);
#endif
    }
    __syncthreads();

    const _Float16* Ksc = Ks[cur];
    const _Float16* Vsc = Vs[cur];

    // S = Q K^T over 4 key tiles: B[d][key] = K[key][d], d contiguous in Ks.
    v8f s[4] = {};
#pragma unroll
    for (int t = 0; t < 4; ++t) {
      const _Float16* kp = Ksc + (size_t)(t * 16 + l16) * 64;
      s[t] = wmma16(qf0, b_frag_h16(kp + 0, half), s[t]);
      s[t] = wmma16(qf1, b_frag_h16(kp + 32, half), s[t]);
    }

    // Online softmax: C-layout row = r + 8*half, uniform across 16-lane halves.
#pragma unroll
    for (int r = 0; r < 8; ++r) {
      float rm = fmaxf(fmaxf(s[0][r], s[1][r]), fmaxf(s[2][r], s[3][r]));
      rm = fmaxf(rm, __shfl_xor(rm, 1));
      rm = fmaxf(rm, __shfl_xor(rm, 2));
      rm = fmaxf(rm, __shfl_xor(rm, 4));
      rm = fmaxf(rm, __shfl_xor(rm, 8));
      const float mn = fmaxf(m[r], rm);
      const float corr = __expf(m[r] - mn);
      float p[4], rs = 0.f;
#pragma unroll
      for (int t = 0; t < 4; ++t) {
        p[t] = __expf(s[t][r] - mn);
        rs += p[t];
      }
      rs += __shfl_xor(rs, 1);
      rs += __shfl_xor(rs, 2);
      rs += __shfl_xor(rs, 4);
      rs += __shfl_xor(rs, 8);
      l[r] = l[r] * corr + rs;
      m[r] = mn;
      o0[r] *= corr; o1[r] *= corr; o2[r] *= corr; o3[r] *= corr;
      // spill P (C layout) into per-wave LDS, re-read as A fragments below
      _Float16* pr = Pw + (r + 8 * half) * 64 + l16;
#pragma unroll
      for (int t = 0; t < 4; ++t) pr[t * 16] = (_Float16)p[t];
    }

    // O += P @ V : B[key][d] = Vs[d][key], key contiguous along Vs rows.
    const v16h pf0 = a_frag_h16(Pw + l16 * 64 + 0, half);   // keys 0..31
    const v16h pf1 = a_frag_h16(Pw + l16 * 64 + 32, half);  // keys 32..63
#pragma unroll
    for (int t = 0; t < 4; ++t) {
      const _Float16* vp = Vsc + (size_t)(t * 16 + l16) * 64;
      v8f* op = (t == 0) ? &o0 : (t == 1) ? &o1 : (t == 2) ? &o2 : &o3;
      *op = wmma16(pf0, b_frag_h16(vp + 0, half), *op);
      *op = wmma16(pf1, b_frag_h16(vp + 32, half), *op);
    }
    __syncthreads();
  }

  const int b = bh / HEADS, h = bh % HEADS;
  v8f oo[4] = {o0, o1, o2, o3};
#pragma unroll
  for (int t = 0; t < 4; ++t) {
    const int e = h * 64 + t * 16 + l16;
    _Float16* op = O + ((size_t)b * NSEQ + q0 + 8 * half) * EMBED + e;
#pragma unroll
    for (int r = 0; r < 8; ++r) {
      const float v = oo[t][r] / (l[r] * 16.0f);  // /= l, /= sqrt(EMBED)
      op[(size_t)r * EMBED] = (_Float16)v;
    }
  }
}

// ---- Stage 3: output projection + bias + residual --------------------------
// grid = (B*N/32), block = 128 (4 waves). out = O @ Wp^T + bp + x  (fp32)
__global__ __launch_bounds__(128) void out_proj_kernel(
    const _Float16* __restrict__ O, const _Float16* __restrict__ wph,
    const float* __restrict__ bp, const float* __restrict__ x,
    float* __restrict__ out) {
  const int lane = threadIdx.x & 31, wid = threadIdx.x >> 5;
  const int half = lane >> 4, l16 = lane & 15;
  const int g0 = blockIdx.x * 32;
  const _Float16* orow0 = O + (size_t)(g0 + l16) * EMBED;
  const _Float16* orow1 = orow0 + (size_t)16 * EMBED;
  const int eb = wid * 64;

  v8f acc[2][4] = {};
#pragma unroll
  for (int k0 = 0; k0 < EMBED; k0 += 32) {
    const v16h a0 = a_frag_h16(orow0 + k0, half);
    const v16h a1 = a_frag_h16(orow1 + k0, half);
#pragma unroll
    for (int t = 0; t < 4; ++t) {
      const v16h bf = b_frag_h16(wph + (size_t)(eb + t * 16 + l16) * EMBED + k0, half);
      acc[0][t] = wmma16(a0, bf, acc[0][t]);
      acc[1][t] = wmma16(a1, bf, acc[1][t]);
    }
  }
#pragma unroll
  for (int mt = 0; mt < 2; ++mt) {
#pragma unroll
    for (int t = 0; t < 4; ++t) {
      const int e = eb + t * 16 + l16;
      const float bb = bp[e];
#pragma unroll
      for (int r = 0; r < 8; ++r) {
        const size_t idx = ((size_t)(g0 + mt * 16 + r + 8 * half)) * EMBED + e;
        out[idx] = acc[mt][t][r] + bb + x[idx];
      }
    }
  }
}

// ---- Host launcher ----------------------------------------------------------
extern "C" void kernel_launch(void* const* d_in, const int* in_sizes, int n_in,
                              void* d_out, int out_size, void* d_ws,
                              size_t ws_size, hipStream_t stream) {
  (void)in_sizes; (void)n_in; (void)out_size; (void)ws_size;
  const float* x  = (const float*)d_in[0];
  const float* wq = (const float*)d_in[1];
  const float* bq = (const float*)d_in[2];
  const float* wk = (const float*)d_in[3];
  const float* bk = (const float*)d_in[4];
  const float* wv = (const float*)d_in[5];
  const float* bv = (const float*)d_in[6];
  const float* wp = (const float*)d_in[7];
  const float* bp = (const float*)d_in[8];
  float* out = (float*)d_out;

  // Workspace carve-up (f16 halves):
  const size_t nX = (size_t)BATCH * NSEQ * EMBED;          // 2,097,152
  const size_t nW = (size_t)EMBED * EMBED;                 // 65,536
  const size_t nH = (size_t)BATCH * HEADS * NSEQ * DHEAD;  // 2,097,152
  _Float16* xh  = (_Float16*)d_ws;
  _Float16* wqh = xh + nX;
  _Float16* wkh = wqh + nW;
  _Float16* wvh = wkh + nW;
  _Float16* wph = wvh + nW;
  _Float16* Q   = wph + nW;
  _Float16* K   = Q + nH;
  _Float16* Vt  = K + nH;
  _Float16* O   = Vt + nH;

  // Stage 0: one-shot f16 conversion of activations + weights.
  cvt_f16_kernel<<<dim3((int)(nX / 8 / 256)), 256, 0, stream>>>(x, xh, (int)(nX / 8));
  cvt_f16_kernel<<<dim3((int)(nW / 8 / 256)), 256, 0, stream>>>(wq, wqh, (int)(nW / 8));
  cvt_f16_kernel<<<dim3((int)(nW / 8 / 256)), 256, 0, stream>>>(wk, wkh, (int)(nW / 8));
  cvt_f16_kernel<<<dim3((int)(nW / 8 / 256)), 256, 0, stream>>>(wv, wvh, (int)(nW / 8));
  cvt_f16_kernel<<<dim3((int)(nW / 8 / 256)), 256, 0, stream>>>(wp, wph, (int)(nW / 8));

  qkv_proj_kernel<<<dim3((BATCH * NSEQ) / 32, 3), 128, 0, stream>>>(
      xh, wqh, bq, wkh, bk, wvh, bv, Q, K, Vt);
  flash_attn_kernel<<<dim3(NSEQ / 64, BATCH * HEADS), 128, 0, stream>>>(Q, K, Vt, O);
  out_proj_kernel<<<dim3((BATCH * NSEQ) / 32), 128, 0, stream>>>(O, wph, bp, x, out);
}
